// VectorQuantizerEMA_19344532701258
// MI455X (gfx1250) — matrix operand order repaired and checked by reading, not compiled, not run on hardware
//
#include <hip/hip_runtime.h>
#include <hip/hip_bf16.h>

typedef __attribute__((ext_vector_type(16))) __bf16 v16bf;
typedef __attribute__((ext_vector_type(8)))  float  v8f;
typedef __attribute__((ext_vector_type(4)))  unsigned int v4u;
typedef __attribute__((ext_vector_type(8)))  int    v8i;
typedef __attribute__((ext_vector_type(4)))  int    v4i;

#define VQ_N 32768
#define VQ_D 256
#define VQ_K 8192
#define TILE_M 128
#define TILE_C 64                        // codes per W tile (halved: 2 WGs/WGP)
#define ZPAD 264                         // LDS row stride in bf16 elems (528 B)
#define ZBUF_BYTES (TILE_M * ZPAD * 2)   // 67584 B: Z tile
#define WBUF_BYTES (TILE_C * ZPAD * 2)   // 33792 B: one W tile buffer

__device__ __forceinline__ unsigned short f2bf(float f) {
  unsigned int u = __float_as_uint(f);
  u += 0x7FFFu + ((u >> 16) & 1u);    // RNE truncate to bf16
  return (unsigned short)(u >> 16);
}
__device__ __forceinline__ unsigned int pack2bf(float a, float b) {
  return (unsigned int)f2bf(a) | ((unsigned int)f2bf(b) << 16);
}

// ---------------------------------------------------------------------------
// TDM: async DMA of one 64x256 bf16 codebook tile (row-major, contiguous in
// global) into LDS with 16B padding after each 512B row (row stride 528B).
// Descriptor per CDNA5 ISA ch.8: 2D tensor, data_size=2B, pad_interval=6
// (128 DWORDs), pad_amount=3 (4 DWORDs). Groups 2/3 zero (<=2D).
// ---------------------------------------------------------------------------
__device__ __forceinline__ void tdm_load_wtile(const unsigned short* Wbf, int c0,
                                               unsigned int lds_off) {
  const unsigned long long ga =
      (unsigned long long)(const void*)(Wbf + (size_t)c0 * VQ_D);
  v4u g0;
  g0.x = 1u;                                             // count=1, user mode
  g0.y = lds_off;                                        // lds_addr [63:32]
  g0.z = (unsigned int)ga;                               // global_addr lo
  g0.w = (unsigned int)((ga >> 32) & 0x01FFFFFFull)      // global_addr [56:32]
         | 0x80000000u;                                  // type=2 ("image")
  const v8i g1 = {
      (int)((1u << 16) | (1u << 20) | (6u << 22) | (3u << 25)),
      // ^ data_size=2B | pad_enable | pad_interval=6 | pad_amount=3
      (int)(256u << 16),   // [63:48]  tensor_dim0 = 256 elems
      (int)(8192u << 16),  // [95:80]  tensor_dim1 = 8192 rows
      (int)(256u << 16),   // [127:112] tile_dim0 = 256 elems
      TILE_C,              // [143:128] tile_dim1 = 64 rows
      256,                 // [191:160] tensor_dim0_stride = 256 elems
      0, 0};
  const v4i gz4 = {0, 0, 0, 0};
  const v8i gz8 = {0, 0, 0, 0, 0, 0, 0, 0};
  __builtin_amdgcn_tensor_load_to_lds(g0, g1, gz4, gz4, gz8, 0);
}

// ---------------------------------------------------------------------------
// Kernel 1: codebook prep — W (f32) -> bf16 workspace + 0.5*||W_k||^2
// ---------------------------------------------------------------------------
__global__ __launch_bounds__(256) void vq_prep(const float* __restrict__ W,
                                               unsigned short* __restrict__ Wbf,
                                               float* __restrict__ wn) {
  const int t = threadIdx.x;
  const int lane = t & 31;
  const int wv = t >> 5;
  const int code = blockIdx.x * 8 + wv;
  const float4* src = (const float4*)(W + (size_t)code * VQ_D);
  unsigned int* dst = (unsigned int*)(Wbf + (size_t)code * VQ_D);
  float ss = 0.f;
#pragma unroll
  for (int p = 0; p < 2; ++p) {
    const int c4 = lane * 2 + p;
    float4 v = src[c4];
    ss += v.x * v.x + v.y * v.y + v.z * v.z + v.w * v.w;
    dst[c4 * 2 + 0] = pack2bf(v.x, v.y);
    dst[c4 * 2 + 1] = pack2bf(v.z, v.w);
  }
#pragma unroll
  for (int off = 16; off >= 1; off >>= 1) ss += __shfl_xor(ss, off, 32);
  if (lane == 0) wn[code] = 0.5f * ss;
}

// ---------------------------------------------------------------------------
// Kernel 2: fused bf16-WMMA GEMM + argmin, TDM double-buffered W tiles
// ---------------------------------------------------------------------------
__global__ __launch_bounds__(256) void vq_main(const float* __restrict__ z,
                                               const float* __restrict__ W,
                                               const unsigned short* __restrict__ Wbf,
                                               const float* __restrict__ wn,
                                               float* __restrict__ outZq,
                                               float* __restrict__ outIdx,
                                               float* __restrict__ partials) {
  extern __shared__ char smem_raw[];
  unsigned short* lds_z = (unsigned short*)smem_raw;              // 128x264 bf16
  unsigned short* lds_w = lds_z + TILE_M * ZPAD;                  // 2 x 64x264 bf16
  float* lds_ws = (float*)(smem_raw + ZBUF_BYTES + 2 * WBUF_BYTES); // 8 f32

  const int t = threadIdx.x;
  const int lane = t & 31;
  const int wv = t >> 5;
  const int m0 = blockIdx.x * TILE_M;
  const int nTiles = VQ_K / TILE_C;   // 128

  // kick off DMA of first two W tiles while we stage Z
  if (wv == 0) {
    tdm_load_wtile(Wbf, 0, ZBUF_BYTES);
    tdm_load_wtile(Wbf, TILE_C, ZBUF_BYTES + WBUF_BYTES);
  }

  // ---- stage Z tile, f32 -> bf16, once per block ----
#pragma unroll
  for (int i = 0; i < 32; ++i) {
    const int idx4 = t + i * 256;            // 8192 float4 total
    const int row = idx4 >> 6;
    const int c4 = idx4 & 63;
    float4 v = ((const float4*)(z + (size_t)(m0 + row) * VQ_D))[c4];
    unsigned int* p = (unsigned int*)(lds_z + row * ZPAD + c4 * 4);
    p[0] = pack2bf(v.x, v.y);
    p[1] = pack2bf(v.z, v.w);
  }

  if (wv == 0) __builtin_amdgcn_s_wait_tensorcnt(1);   // tile 0 landed
  __syncthreads();

  const int nn = lane & 15;   // column in 16x16 tile; also A-matrix row
  const int hh = lane >> 4;   // half-wave select

  float best[8];
  unsigned int bidx[8];
#pragma unroll
  for (int r = 0; r < 8; ++r) { best[r] = 3.4e38f; bidx[r] = 0u; }

  const unsigned short* zrow = lds_z + (wv * 16 + nn) * ZPAD;

  for (int kt = 0; kt < nTiles; ++kt) {
    const int c0 = kt * TILE_C;
    const int cur = kt & 1;
    const unsigned short* wtile = lds_w + cur * (TILE_C * ZPAD);

    v8f acc[4];
    const v8f vzero = {0.f, 0.f, 0.f, 0.f, 0.f, 0.f, 0.f, 0.f};
#pragma unroll
    for (int j = 0; j < 4; ++j) acc[j] = vzero;

#pragma unroll
    for (int dc = 0; dc < 8; ++dc) {         // reduction over D in chunks of 32
      const int d0 = dc * 32;
      union { uint4 q[2]; v16bf v; } A;
      const uint4* zp = (const uint4*)(zrow + d0);
      A.q[0] = zp[hh];                       // K = 8h .. 8h+7
      A.q[1] = zp[2 + hh];                   // K = 16+8h .. 16+8h+7
      union { uint4 q[2]; v16bf v; } B[4];
#pragma unroll
      for (int j = 0; j < 4; ++j) {
        const uint4* wp = (const uint4*)(wtile + (j * 16 + nn) * ZPAD + d0);
        B[j].q[0] = wp[hh];                  // K = 8h .. 8h+7
        B[j].q[1] = wp[2 + hh];              // K = 16+8h .. 16+8h+7
      }
#pragma unroll
      for (int j = 0; j < 4; ++j) {
        acc[j] = __builtin_amdgcn_wmma_f32_16x16x32_bf16(
            false, A.v, false, B[j].v, (short)0, acc[j], false, false);
      }
    }

    // ---- fused epilogue: score = 0.5||W||^2 - z.W ; running argmin ----
#pragma unroll
    for (int j = 0; j < 4; ++j) {
      const float wnv = wn[c0 + j * 16 + nn];          // L2-hot, 256B/tile
      const unsigned int code = (unsigned int)(c0 + j * 16 + nn);
#pragma unroll
      for (int r = 0; r < 8; ++r) {
        const float s = wnv - acc[j][r];
        if (s < best[r] || (s == best[r] && code < bidx[r])) { best[r] = s; bidx[r] = code; }
      }
    }

    __syncthreads();                          // all waves done reading buf[cur]
    if (wv == 0) {
      if (kt + 2 < nTiles) {
        tdm_load_wtile(Wbf, (kt + 2) * TILE_C,
                       ZBUF_BYTES + (unsigned int)cur * WBUF_BYTES);
        __builtin_amdgcn_s_wait_tensorcnt(1); // tile kt+1 complete (in-order)
      } else {
        __builtin_amdgcn_s_wait_tensorcnt(0);
      }
    }
    __syncthreads();                          // buf[cur^1] ready for all waves
  }

  // ---- cross-lane argmin within each 16-lane half ----
#pragma unroll
  for (int off = 1; off <= 8; off <<= 1) {
#pragma unroll
    for (int r = 0; r < 8; ++r) {
      const float ov = __shfl_xor(best[r], off, 32);
      const unsigned int oi = (unsigned int)__shfl_xor((int)bidx[r], off, 32);
      if (ov < best[r] || (ov == best[r] && oi < bidx[r])) { best[r] = ov; bidx[r] = oi; }
    }
  }

  if (nn == 0) {
#pragma unroll
    for (int r = 0; r < 8; ++r) {
      const int row = m0 + wv * 16 + r + 8 * hh;
      outIdx[row] = (float)bidx[r];
    }
  }

  // ---- gather z_q from exact f32 codebook + loss partial ----
  float dsum = 0.f;
#pragma unroll
  for (int rr = 0; rr < 16; ++rr) {
    const int r = rr & 7;
    const int src_lane = (rr >> 3) * 16;
    const unsigned int code = (unsigned int)__shfl((int)bidx[r], src_lane, 32);
    const size_t row = (size_t)(m0 + wv * 16 + rr);
    const float4* wsrc = (const float4*)(W + (size_t)code * VQ_D);
    const float4* zsrc = (const float4*)(z + row * VQ_D);
    float4* dst = (float4*)(outZq + row * VQ_D);
#pragma unroll
    for (int p = 0; p < 2; ++p) {
      const int c4 = lane * 2 + p;
      float4 wq = wsrc[c4];
      float4 zv = zsrc[c4];
      dst[c4] = wq;
      const float dx = wq.x - zv.x, dy = wq.y - zv.y;
      const float dz2 = wq.z - zv.z, dw = wq.w - zv.w;
      dsum += dx * dx + dy * dy + dz2 * dz2 + dw * dw;
    }
  }
#pragma unroll
  for (int off = 16; off >= 1; off >>= 1) dsum += __shfl_xor(dsum, off, 32);
  if (lane == 0) lds_ws[wv] = dsum;
  __syncthreads();
  if (t == 0) {
    float s = 0.f;
#pragma unroll
    for (int w2 = 0; w2 < 8; ++w2) s += lds_ws[w2];    // fixed order: deterministic
    partials[blockIdx.x] = s;
  }
}

// ---------------------------------------------------------------------------
// Kernel 3: deterministic reduction of 256 block partials -> diff scalar
// ---------------------------------------------------------------------------
__global__ __launch_bounds__(256) void vq_finalize(const float* __restrict__ partials,
                                                   float* __restrict__ diffout) {
  __shared__ float s[256];
  const int t = threadIdx.x;
  s[t] = partials[t];
  __syncthreads();
  for (int st = 128; st > 0; st >>= 1) {
    if (t < st) s[t] += s[t + st];
    __syncthreads();
  }
  if (t == 0) diffout[0] = 2.0f * s[0] / (float)((size_t)VQ_N * VQ_D);
}

// ---------------------------------------------------------------------------
extern "C" void kernel_launch(void* const* d_in, const int* in_sizes, int n_in,
                              void* d_out, int out_size, void* d_ws, size_t ws_size,
                              hipStream_t stream) {
  (void)in_sizes; (void)n_in; (void)out_size; (void)ws_size;
  const float* z = (const float*)d_in[0];
  const float* W = (const float*)d_in[1];
  float* out = (float*)d_out;
  char* ws = (char*)d_ws;
  unsigned short* Wbf = (unsigned short*)ws;                                 // 4 MB
  float* wn = (float*)(ws + (size_t)VQ_K * VQ_D * sizeof(unsigned short));   // 32 KB
  float* partials = wn + VQ_K;                                               // 1 KB

  const size_t smem = (size_t)ZBUF_BYTES + 2 * WBUF_BYTES + 8 * sizeof(float); // ~135 KB
  (void)hipFuncSetAttribute((const void*)vq_main,
                            hipFuncAttributeMaxDynamicSharedMemorySize, (int)smem);

  vq_prep<<<VQ_K / 8, 256, 0, stream>>>(W, Wbf, wn);
  vq_main<<<VQ_N / TILE_M, 256, smem, stream>>>(z, W, Wbf, wn,
                                                out, out + (size_t)VQ_N * VQ_D, partials);
  vq_finalize<<<1, 256, 0, stream>>>(partials, out + (size_t)VQ_N * VQ_D + VQ_N);
}
